// LSTM_M_24962349925002
// MI455X (gfx1250) — matrix operand order repaired and checked by reading, not compiled, not run on hardware
//
#include <hip/hip_runtime.h>
#include <hip/hip_bf16.h>
#include <stdint.h>

// ---------------------------------------------------------------------------
// LSTM  (B=256, T=512, C=512, U=512)  for MI455X (gfx1250, wave32, WMMA)
//
//  Kernel 1: pack wx/wh (f32 row-major) -> bf16 WMMA B-fragment layout
//  Kernel 2: Z = X @ wx + b  (bf16 WMMA, f32 accum), stored as bf16 D-frags
//            8 waves <-> 8 timesteps; A-frags register-resident.
//  Kernel 3: sequential recurrence, 16 WGs x 512 threads (16 waves ->
//            4 waves/SIMD to hide B-frag L2 latency under the WMMA chain).
//            h in LDS, c in registers; kb-outer / gate-inner WMMA loop
//            (4 independent accumulator chains per wave).
// ---------------------------------------------------------------------------

typedef __attribute__((ext_vector_type(16))) __bf16 v16bf;
typedef __attribute__((ext_vector_type(8)))  float  v8f;

#define Bn 256
#define Tn 512
#define Cn 512
#define Un 512
#define Gn 2048          // 4*U
#define NT 128           // Gn / 16  (N tiles)
#define KB 16            // Cn / 32  (K blocks of 32)
#define HSTRIDE 516      // LDS row pitch for h (bank spread, 16B aligned)

// ---------------------------------------------------------------------------
// Kernel 1: pack wx and wh into bf16 B-fragment order
//   layout: ((kb*NT + nt)*32 + lane)*16 + j ;  n = nt*16 + lane%16,
//   K = 8*(lane/16) + (j&7) + 16*(j/8)   (ISA 16-bit table order, matches A frags)
// ---------------------------------------------------------------------------
__global__ void pack_w_kernel(const float* __restrict__ wx,
                              const float* __restrict__ wh,
                              __bf16* __restrict__ wxP,
                              __bf16* __restrict__ whP) {
    int idx  = blockIdx.x * blockDim.x + threadIdx.x;   // 0 .. 2^20-1
    int j    = idx & 15;
    int lane = (idx >> 4) & 31;
    int nt   = (idx >> 9) & (NT - 1);
    int kb   = idx >> 16;
    if (kb >= KB) return;
    int K = 8 * (lane >> 4) + (j & 7) + 16 * (j >> 3);
    int n = nt * 16 + (lane & 15);
    int src = (kb * 32 + K) * Gn + n;
    wxP[idx] = (__bf16)wx[src];
    whP[idx] = (__bf16)wh[src];
}

// ---------------------------------------------------------------------------
// Kernel 2: Z[wgm, t, nt, lane, 0..7] = bf16( X @ wx + b )  as D-fragments
//   grid = (16, Tn/8), block = 256; wave w handles t = t0 + w.
// ---------------------------------------------------------------------------
__global__ void __launch_bounds__(256)
precompute_z_kernel(const float* __restrict__ x,
                    const float* __restrict__ bias,
                    const __bf16* __restrict__ wxP,
                    __bf16* __restrict__ Z) {
    const int wgm  = blockIdx.x;              // batch tile 0..15
    const int wave = threadIdx.x >> 5;
    const int lane = threadIdx.x & 31;
    const int lm   = lane & 15;
    const int lh   = lane >> 4;
    const int t    = blockIdx.y * 8 + wave;   // this wave's timestep
    const int b    = wgm * 16 + lm;           // this lane's batch row (A: M)

    const float* xr = x + ((size_t)b * Tn + t) * Cn;

    // Build A-fragments in registers.  ISA 7.12.2 16-bit A 16x32 per-lane map:
    // lanes 0-15: K = {0..7, 16..23}; lanes 16-31: K = {8..15, 24..31}
    v16bf afr[KB];
#pragma unroll
    for (int kb = 0; kb < KB; ++kb) {
        const float* b0p = xr + kb * 32 + 8 * lh;
        const float* b1p = xr + kb * 32 + 16 + 8 * lh;
        float r0[8], r1[8];
#pragma unroll
        for (int q = 0; q < 8; ++q) { r0[q] = b0p[q]; r1[q] = b1p[q]; }
        v16bf a;
#pragma unroll
        for (int q = 0; q < 8; ++q) { a[q] = (__bf16)r0[q]; a[8 + q] = (__bf16)r1[q]; }
        afr[kb] = a;
    }

    const __bf16* wbase = wxP + (size_t)lane * 16;   // + (kb*NT + nt)*512

#pragma unroll 1
    for (int nt = 0; nt < NT; ++nt) {
        asm volatile("" ::: "memory");        // keep B-frag loads inside loop
        const float bv = bias[nt * 16 + lm];
        v8f acc = {bv, bv, bv, bv, bv, bv, bv, bv};
#pragma unroll
        for (int kb = 0; kb < KB; ++kb) {
            v16bf bf = *(const v16bf*)(wbase + ((size_t)kb * NT + nt) * 512);
            acc = __builtin_amdgcn_wmma_f32_16x16x32_bf16(
                    false, afr[kb], false, bf, (short)0, acc, false, false);
        }
        union { __bf16 h[8]; uint4 u; } ob;
#pragma unroll
        for (int r = 0; r < 8; ++r) ob.h[r] = (__bf16)acc[r];
        size_t zo = ((((size_t)wgm * Tn + t) * NT + nt) * 32 + lane) * 8;
        *(uint4*)(Z + zo) = ob.u;
    }
}

// ---------------------------------------------------------------------------
// Kernel 3: sequential recurrence.  grid = 16, block = 512 (16 waves).
//   wave w owns u-column tiles j = 2w, 2w+1 (i/f/o/u at nt = j + g*32).
// ---------------------------------------------------------------------------
__global__ void __launch_bounds__(512)
lstm_rec_kernel(const float* __restrict__ dones,
                const __bf16* __restrict__ whP,
                const __bf16* __restrict__ Z,
                float* __restrict__ xs,       // [B,T,U]
                float* __restrict__ sout) {   // [B,2U]
    __shared__ float hls[16 * HSTRIDE];
    __shared__ float dk[16];

    const int wg   = blockIdx.x;       // batch tile 0..15
    const int tid  = threadIdx.x;
    const int wave = tid >> 5;         // 0..15
    const int lane = tid & 31;
    const int lm   = lane & 15;
    const int lh   = lane >> 4;
    const int b0   = wg * 16;

    for (int e = tid; e < 16 * HSTRIDE; e += 512) hls[e] = 0.0f;

    v8f creg[2];
#pragma unroll
    for (int jj = 0; jj < 2; ++jj) creg[jj] = (v8f){0,0,0,0,0,0,0,0};

    const __bf16* wbase = whP + (size_t)lane * 16;   // + (kb*NT + nt)*512

    __syncthreads();

#pragma unroll 1
    for (int t = 0; t < Tn; ++t) {
        asm volatile("" ::: "memory");   // keep whP/Z streaming inside t loop

        // ---------- read phase: stream out h(t-1), build A-frags of h*keep
        if (t > 0) {
            int row = tid >> 5, seg = tid & 31;
            const float4* src = (const float4*)(hls + row * HSTRIDE + seg * 16);
            float4* dst = (float4*)(xs + ((size_t)(b0 + row) * Tn + (t - 1)) * Un + seg * 16);
#pragma unroll
            for (int q = 0; q < 4; ++q) dst[q] = src[q];
        }
        if (tid < 16) dk[tid] = 1.0f - dones[(size_t)(b0 + tid) * Tn + t];

        const float keepl = 1.0f - dones[(size_t)(b0 + lm) * Tn + t];
        v16bf afr[KB];
#pragma unroll
        for (int kb = 0; kb < KB; ++kb) {
            const float* hr = hls + lm * HSTRIDE + kb * 32 + 8 * lh;
            v16bf a;
#pragma unroll
            for (int q = 0; q < 8; ++q) a[q]     = (__bf16)(hr[q]      * keepl);
#pragma unroll
            for (int q = 0; q < 8; ++q) a[8 + q] = (__bf16)(hr[16 + q] * keepl);
            afr[kb] = a;
        }
        __syncthreads();

        // ---------- compute phase: z = Z + h@wh, gates, state update
#pragma unroll
        for (int jj = 0; jj < 2; ++jj) {
            const int j = wave * 2 + jj;

            // init 4 gate accumulators from precomputed Z (one load clause)
            v8f zg[4];
#pragma unroll
            for (int g = 0; g < 4; ++g) {
                const int nt = j + g * 32;
                size_t zo = ((((size_t)wg * Tn + t) * NT + nt) * 32 + lane) * 8;
                uint4 raw = *(const uint4*)(Z + zo);
                const __bf16* zb = (const __bf16*)&raw;
#pragma unroll
                for (int r = 0; r < 8; ++r) zg[g][r] = (float)zb[r];
            }

            // kb-outer / gate-inner: one clause of 8 b128 loads feeds
            // 4 independent back-to-back WMMAs (4 accumulator chains);
            // 4 waves/SIMD interleave to keep the matrix pipe busy.
#pragma unroll
            for (int kb = 0; kb < KB; ++kb) {
                v16bf bf[4];
#pragma unroll
                for (int g = 0; g < 4; ++g)
                    bf[g] = *(const v16bf*)(wbase +
                              ((size_t)kb * NT + (j + g * 32)) * 512);
#pragma unroll
                for (int g = 0; g < 4; ++g)
                    zg[g] = __builtin_amdgcn_wmma_f32_16x16x32_bf16(
                              false, afr[kb], false, bf[g], (short)0, zg[g], false, false);
            }

            // gates + state update
            v8f c = creg[jj];
#pragma unroll
            for (int r = 0; r < 8; ++r) {
                float keep = dk[r + 8 * lh];
                float ig = 1.0f / (1.0f + __expf(-zg[0][r]));
                float fg = 1.0f / (1.0f + __expf(-zg[1][r]));
                float og = 1.0f / (1.0f + __expf(-zg[2][r]));
                float ug = tanhf(zg[3][r]);
                float cv = fg * (c[r] * keep) + ig * ug;
                c[r] = cv;
                hls[(r + 8 * lh) * HSTRIDE + j * 16 + lm] = og * tanhf(cv);
            }
            creg[jj] = c;
        }
        __syncthreads();
    }

    // ---------- epilogue: xs[t=T-1], final state s = [c, h]
    {
        int row = tid >> 5, seg = tid & 31;
        const float4* src = (const float4*)(hls + row * HSTRIDE + seg * 16);
        float4* dst = (float4*)(xs + ((size_t)(b0 + row) * Tn + (Tn - 1)) * Un + seg * 16);
        float4* hdst = (float4*)(sout + (size_t)(b0 + row) * (2 * Un) + Un + seg * 16);
#pragma unroll
        for (int q = 0; q < 4; ++q) { dst[q] = src[q]; hdst[q] = src[q]; }
    }
#pragma unroll
    for (int jj = 0; jj < 2; ++jj) {
        const int j = wave * 2 + jj;
#pragma unroll
        for (int r = 0; r < 8; ++r) {
            int b = b0 + r + 8 * lh;
            sout[(size_t)b * (2 * Un) + j * 16 + lm] = creg[jj][r];
        }
    }
}

// ---------------------------------------------------------------------------
// Host launcher.  d_in order: x, dones, wx, wh, b
// ws layout: [0,2MB) wxPack | [2MB,4MB) whPack | [4MB, 4MB+512MB) Z (bf16)
// ---------------------------------------------------------------------------
extern "C" void kernel_launch(void* const* d_in, const int* in_sizes, int n_in,
                              void* d_out, int out_size, void* d_ws, size_t ws_size,
                              hipStream_t stream) {
    const float* x     = (const float*)d_in[0];
    const float* dones = (const float*)d_in[1];
    const float* wx    = (const float*)d_in[2];
    const float* wh    = (const float*)d_in[3];
    const float* bias  = (const float*)d_in[4];

    const size_t packElems = (size_t)KB * NT * 32 * 16;          // 1M bf16 = 2MB
    __bf16* wxP = (__bf16*)d_ws;
    __bf16* whP = wxP + packElems;
    __bf16* Z   = whP + packElems;                               // 512MB

    float* xs   = (float*)d_out;                                 // [B,T,U]
    float* sout = xs + (size_t)Bn * Tn * Un;                     // [B,2U]

    pack_w_kernel<<<4096, 256, 0, stream>>>(wx, wh, wxP, whP);

    dim3 g2(16, Tn / 8);
    precompute_z_kernel<<<g2, 256, 0, stream>>>(x, bias, wxP, Z);

    lstm_rec_kernel<<<16, 512, 0, stream>>>(dones, whP, Z, xs, sout);
}
